// ViT_72335839199699
// MI455X (gfx1250) — compile-verified, matrix operand order checked by reading, not run on hardware
//
#include <hip/hip_runtime.h>

// ---------------- types ----------------
typedef __attribute__((ext_vector_type(16))) __bf16 v16bf;
typedef __attribute__((ext_vector_type(8)))  __bf16 v8bf;
typedef __attribute__((ext_vector_type(4)))  __bf16 v4bf;
typedef __attribute__((ext_vector_type(8)))  float  v8f;
typedef __attribute__((ext_vector_type(4)))  float  v4f;

// ---------------- problem constants ----------------
#define C_CH    3
#define IMG     224
#define PBLK    14          // patches per row
#define PS      16          // patch size
#define NPATCH  196         // 14*14
#define KDIM    768         // patch dim = 3*16*16
#define HID     768
#define SEQ     197

// ---------------- tiling ----------------
#define MT 64               // rows (patches) per workgroup
#define NT 128              // hidden cols per workgroup
#define KT 32               // K per step (one bf16 WMMA K)
#define KP (KT + 8)         // padded LDS row (80 B, 16B-aligned, conflict-free)

__device__ __forceinline__ v16bf cat8(v8bf a, v8bf b) {
    return __builtin_shufflevector(a, b, 0,1,2,3,4,5,6,7,8,9,10,11,12,13,14,15);
}

__device__ __forceinline__ v16bf ldfrag(const __bf16* row, int g) {
    // ISA 16-bit A/B layout: lane group g reads K-chunks [g*8, g*8+8) and [16+g*8, ...)
    return cat8(*(const v8bf*)(row + g * 8), *(const v8bf*)(row + g * 8 + 16));
}

#define WMMA_BF16(A, B, C) \
    __builtin_amdgcn_wmma_f32_16x16x32_bf16(false, (A), false, (B), (short)0, (C), false, false)

__global__ __launch_bounds__(256) void vit_patch_embed(
    const float* __restrict__ images,   // [n,3,224,224]
    const float* __restrict__ Wl,       // [768,768] (row h, col d)
    const float* __restrict__ bl,       // [768]
    float* __restrict__ out)            // [n,197,768]
{
    __shared__ __align__(16) __bf16 Ahi[MT][KP];
    __shared__ __align__(16) __bf16 Alo[MT][KP];
    __shared__ __align__(16) __bf16 Bhi[NT][KP];
    __shared__ __align__(16) __bf16 Blo[NT][KP];

    const int tid   = (int)threadIdx.x;
    const int mbase = (int)blockIdx.y * MT;
    const int nbase = (int)blockIdx.x * NT;

    // ---- A-loader mapping (fused patchify gather), constant across K loop ----
    // thread -> (local patch rows ml0 and ml0+32, image-row segment seg, float4 slot aq)
    const int aq  = tid & 3;            // float4 within 16-float segment
    const int seg = (tid >> 2) & 1;     // which of the two ph rows in this K chunk
    const int ml0 = tid >> 3;           // local A row 0..31 (also handles +32)
    const float* abase[2];
    #pragma unroll
    for (int h = 0; h < 2; ++h) {
        const int m   = mbase + ml0 + 32 * h;
        const int img = m / NPATCH;
        const int p   = m - img * NPATCH;
        const int pi  = p / PBLK;
        const int pj  = p - pi * PBLK;
        abase[h] = images + (size_t)img * (C_CH * IMG * IMG)
                          + (size_t)(pi * PS + seg) * IMG
                          + pj * PS + aq * 4;
    }

    // ---- B-loader mapping: B[k][n] = Wl[n][k]; rows of Wl are contiguous K ----
    const int bq = tid & 7;             // float4 within 32-float K row
    const int bn = tid >> 3;            // 0..31, handles rows bn + 32*i
    const float* bbase = Wl + (size_t)(nbase + bn) * KDIM + bq * 4;

    // ---- wave mapping: 8 waves -> 2 (M) x 4 (N); each wave owns 32x32 ----
    const int lane = tid & 31;
    const int wave = tid >> 5;
    const int wm   = wave & 1;          // 32-row M subtile pair
    const int wn   = wave >> 1;         // 32-col N subtile pair
    const int g    = lane >> 4;         // lane group (selects K-halves / M-halves)
    const int r    = lane & 15;         // row-in-tile (A) / col-in-tile (B,D)

    v8f acc[2][2] = {};                 // [mi][ni]

    #pragma unroll 1
    for (int step = 0; step < KDIM / KT; ++step) {
        const int k0 = step * KT;
        const int c  = k0 >> 8;          // channel
        const int h0 = (k0 >> 4) & 15;   // first ph row of this chunk (even)

        // ---- global loads (16B vectors) ----
        v4f av[2];
        #pragma unroll
        for (int h = 0; h < 2; ++h)
            av[h] = *(const v4f*)(abase[h] + c * (IMG * IMG) + h0 * IMG);
        v4f bv[4];
        #pragma unroll
        for (int i = 0; i < 4; ++i)
            bv[i] = *(const v4f*)(bbase + k0 + (size_t)(32 * i) * KDIM);

        __syncthreads();   // previous iteration's fragment reads are done

        // ---- fp32 -> bf16 hi/lo split, stage into LDS ----
        const int kl = seg * 16 + aq * 4;
        #pragma unroll
        for (int h = 0; h < 2; ++h) {
            v4bf hi, lo;
            #pragma unroll
            for (int j = 0; j < 4; ++j) {
                const float x = av[h][j];
                const __bf16 hb = (__bf16)x;         // RNE
                hi[j] = hb;
                lo[j] = (__bf16)(x - (float)hb);     // exact residual, then RNE
            }
            *(v4bf*)&Ahi[ml0 + 32 * h][kl] = hi;
            *(v4bf*)&Alo[ml0 + 32 * h][kl] = lo;
        }
        #pragma unroll
        for (int i = 0; i < 4; ++i) {
            v4bf hi, lo;
            #pragma unroll
            for (int j = 0; j < 4; ++j) {
                const float x = bv[i][j];
                const __bf16 hb = (__bf16)x;
                hi[j] = hb;
                lo[j] = (__bf16)(x - (float)hb);
            }
            *(v4bf*)&Bhi[bn + 32 * i][bq * 4] = hi;
            *(v4bf*)&Blo[bn + 32 * i][bq * 4] = lo;
        }
        __syncthreads();

        // ---- fragment loads (two b128 each; ISA 16-bit A/B layout) ----
        v16bf ah[2], al[2], bh[2], blv[2];
        #pragma unroll
        for (int mi = 0; mi < 2; ++mi) {
            const int ar = wm * 32 + mi * 16 + r;
            ah[mi] = ldfrag(&Ahi[ar][0], g);
            al[mi] = ldfrag(&Alo[ar][0], g);
        }
        #pragma unroll
        for (int ni = 0; ni < 2; ++ni) {
            const int br = wn * 32 + ni * 16 + r;
            bh[ni]  = ldfrag(&Bhi[br][0], g);
            blv[ni] = ldfrag(&Blo[br][0], g);
        }

        // ---- bf16x3 split-precision MMA: hi*lo + lo*hi + hi*hi ----
        #pragma unroll
        for (int mi = 0; mi < 2; ++mi)
            #pragma unroll
            for (int ni = 0; ni < 2; ++ni) {
                acc[mi][ni] = WMMA_BF16(ah[mi], blv[ni], acc[mi][ni]);
                acc[mi][ni] = WMMA_BF16(al[mi], bh[ni],  acc[mi][ni]);
                acc[mi][ni] = WMMA_BF16(ah[mi], bh[ni],  acc[mi][ni]);
            }
    }

    // ---- epilogue: bias add + scatter into [img, 1+p, h] ----
    // D layout: lane(g,r), VGPR v -> element (M = g*8+v, N = r)
    const int ncol0 = nbase + wn * 32 + r;
    const int ncol1 = ncol0 + 16;
    const float b0 = bl[ncol0];
    const float b1 = bl[ncol1];
    #pragma unroll
    for (int mi = 0; mi < 2; ++mi) {
        const int rowbase = mbase + wm * 32 + mi * 16 + g * 8;
        int img = rowbase / NPATCH;            // one division per subtile
        int p   = rowbase - img * NPATCH;
        #pragma unroll
        for (int v = 0; v < 8; ++v) {
            float* orow = out + ((size_t)img * SEQ + 1 + p) * HID;
            orow[ncol0] = acc[mi][0][v] + b0;
            orow[ncol1] = acc[mi][1][v] + b1;
            if (++p == NPATCH) { p = 0; ++img; }   // carry instead of div
        }
    }
}

__global__ __launch_bounds__(256) void vit_cls_rows(
    const float* __restrict__ cls, float* __restrict__ out, int n_img)
{
    const int i = (int)blockIdx.x * 256 + (int)threadIdx.x;   // [0, n_img*768)
    if (i >= n_img * HID) return;
    const int n = i / HID;
    const int h = i - n * HID;
    out[(size_t)n * SEQ * HID + h] = cls[h];
}

extern "C" void kernel_launch(void* const* d_in, const int* in_sizes, int n_in,
                              void* d_out, int out_size, void* d_ws, size_t ws_size,
                              hipStream_t stream) {
    const float* images = (const float*)d_in[0];
    const float* Wl     = (const float*)d_in[1];
    const float* bl     = (const float*)d_in[2];
    const float* cls    = (const float*)d_in[3];
    float* out          = (float*)d_out;

    const int n_img = in_sizes[0] / (C_CH * IMG * IMG);     // 128
    const int Mrows = n_img * NPATCH;                       // 25088

    dim3 grid(HID / NT, Mrows / MT);                        // (6, 392)
    vit_patch_embed<<<grid, 256, 0, stream>>>(images, Wl, bl, out);

    const int cls_elems = n_img * HID;
    vit_cls_rows<<<(cls_elems + 255) / 256, 256, 0, stream>>>(cls, out, n_img);
}
